// RoBiTEBase_4973572129051
// MI455X (gfx1250) — matrix operand order repaired and checked by reading, not compiled, not run on hardware
//
#include <hip/hip_runtime.h>
#include <hip/hip_bf16.h>

typedef _Float16 v16h __attribute__((ext_vector_type(16)));
typedef _Float16 v8h  __attribute__((ext_vector_type(8)));
typedef float    v8f  __attribute__((ext_vector_type(8)));
typedef float    v4f  __attribute__((ext_vector_type(4)));

// Async global->LDS 16B copy (CDNA5, ASYNCcnt-tracked). LDS address is the
// low 32 bits of the flat shared pointer (ISA: LDS aperture addr = addr[31:0]).
__device__ __forceinline__ void async_copy_b128(const void* lds_dst, const void* gsrc)
{
    unsigned lofs = (unsigned)(uintptr_t)lds_dst;
    unsigned long long g = (unsigned long long)(uintptr_t)gsrc;
    asm volatile("global_load_async_to_lds_b128 %0, %1, off"
                 :: "v"(lofs), "v"(g) : "memory");
}
__device__ __forceinline__ void async_wait0()
{
    asm volatile("s_wait_asynccnt 0x0" ::: "memory");
}

// ---------------------------------------------------------------------------
// RMSNorm (fp32 in) -> f16 out.  One block per row of D elements.
// ---------------------------------------------------------------------------
__global__ __launch_bounds__(256)
void rmsnorm_cast_kernel(const float* __restrict__ x, const float* __restrict__ w,
                         _Float16* __restrict__ out, int D)
{
    int row = blockIdx.x;
    const float* xr = x + (size_t)row * D;
    float ss = 0.f;
    for (int i = threadIdx.x; i < D; i += 256) { float v = xr[i]; ss += v * v; }
    #pragma unroll
    for (int off = 16; off > 0; off >>= 1) ss += __shfl_xor(ss, off, 32);
    __shared__ float wsum[8];
    int wave = threadIdx.x >> 5, lane = threadIdx.x & 31;
    if (lane == 0) wsum[wave] = ss;
    __syncthreads();
    float tot = 0.f;
    #pragma unroll
    for (int i = 0; i < 8; ++i) tot += wsum[i];
    float scale = rsqrtf(tot / (float)D + 1e-12f);
    for (int i = threadIdx.x; i < D; i += 256)
        out[(size_t)row * D + i] = (_Float16)(xr[i] * scale * w[i]);
}

// ---------------------------------------------------------------------------
// WMMA GEMM:  C[M,N] = A[M,K](f16) * W[N,K](f32, converted to f16)^T
// Block tile 128x64, K-step 32.  8 waves, each wave owns a 32x32 (2x2 WMMA).
// A tile staged with GLOBAL_LOAD_ASYNC_TO_LDS_B128; W converted in-register.
// mode 0: outH = C (f16) | mode 1: outF = resid + C (fp32) | mode 2: silu f16
// ---------------------------------------------------------------------------
#define BM 128
#define BN 64
#define BK 32
#define LDA 40   // padded row stride in halves (16B-aligned)
#define LDB 40

__global__ __launch_bounds__(256)
void gemm_wmma_kernel(const _Float16* __restrict__ A, const float* __restrict__ W,
                      const float* __restrict__ resid, float* __restrict__ outF,
                      _Float16* __restrict__ outH, int M, int N, int K, int mode)
{
    __shared__ __align__(16) _Float16 sA[BM * LDA];
    __shared__ __align__(16) _Float16 sB[BN * LDB];

    int tid  = threadIdx.x;
    int lane = tid & 31;
    int wave = tid >> 5;
    int la = lane & 15, lh = lane >> 4;

    int m0 = blockIdx.y * BM;
    int n0 = blockIdx.x * BN;
    int wm = (wave >> 1) * 32;   // 4 waves along M
    int wn = (wave & 1) * 32;    // 2 waves along N

    v8f acc[2][2] = {};

    for (int k0 = 0; k0 < K; k0 += BK) {
        __syncthreads();
        // ---- A tile (128x32 halves): async copy, 2 x 16B chunks per thread ----
        #pragma unroll
        for (int it = 0; it < 2; ++it) {
            int idx = tid + it * 256;
            int r  = idx >> 2;
            int ck = (idx & 3) * 8;
            async_copy_b128(sA + r * LDA + ck,
                            A + (size_t)(m0 + r) * K + k0 + ck);
        }
        // ---- B tile: W fp32 (64x32) -> f16 in LDS (needs convert; VGPR path) ----
        {
            int r  = tid >> 2;
            int ck = (tid & 3) * 8;
            const float* wp = W + (size_t)(n0 + r) * K + k0 + ck;
            v4f w0 = *(const v4f*)wp;
            v4f w1 = *(const v4f*)(wp + 4);
            v8h bh;
            #pragma unroll
            for (int i = 0; i < 4; ++i) { bh[i] = (_Float16)w0[i]; bh[i + 4] = (_Float16)w1[i]; }
            *(v8h*)(sB + r * LDB + ck) = bh;
        }
        // prefetch next K tile toward the WGP while copies are in flight
        if (k0 + BK < K) {
            __builtin_prefetch(A + (size_t)(m0 + (tid >> 1)) * K + k0 + BK, 0, 3);
            __builtin_prefetch(W + (size_t)(n0 + (tid & 63)) * K + k0 + BK, 0, 3);
        }
        async_wait0();
        __syncthreads();

        // ---- build fragments (16-bit A 16x32 layout / B 32x16 layout) ----
        v16h af[2], bf[2];
        #pragma unroll
        for (int mi = 0; mi < 2; ++mi) {
            const _Float16* p = sA + (wm + mi * 16 + la) * LDA + lh * 8;
            v8h lo = *(const v8h*)p;
            v8h hi = *(const v8h*)(p + 16);
            #pragma unroll
            for (int i = 0; i < 8; ++i) { af[mi][i] = lo[i]; af[mi][i + 8] = hi[i]; }
        }
        #pragma unroll
        for (int ni = 0; ni < 2; ++ni) {
            const _Float16* p = sB + (wn + ni * 16 + la) * LDB + lh * 8;
            v8h lo = *(const v8h*)p;
            v8h hi = *(const v8h*)(p + 16);
            #pragma unroll
            for (int i = 0; i < 8; ++i) { bf[ni][i] = lo[i]; bf[ni][i + 8] = hi[i]; }
        }
        #pragma unroll
        for (int mi = 0; mi < 2; ++mi)
            #pragma unroll
            for (int ni = 0; ni < 2; ++ni)
                acc[mi][ni] = __builtin_amdgcn_wmma_f32_16x16x32_f16(
                    false, af[mi], false, bf[ni], (short)0, acc[mi][ni], false, false);
    }

    // ---- epilogue: C layout -> row = i + 8*lh, col = la ----
    #pragma unroll
    for (int mi = 0; mi < 2; ++mi) {
        #pragma unroll
        for (int ni = 0; ni < 2; ++ni) {
            int rbase = m0 + wm + mi * 16 + lh * 8;
            int c     = n0 + wn + ni * 16 + la;
            #pragma unroll
            for (int i = 0; i < 8; ++i) {
                float vv = acc[mi][ni][i];
                size_t off = (size_t)(rbase + i) * N + c;
                if (mode == 0)      outH[off] = (_Float16)vv;
                else if (mode == 1) outF[off] = resid[off] + vv;
                else                outH[off] = (_Float16)(vv / (1.f + __expf(-vv)));
            }
        }
    }
}

// ---------------------------------------------------------------------------
// Flash attention (f16 QKV in [B,S,NH,HD], f16 out in [B,S,D]).
// One block = 8 waves = 128 query rows of one (batch, head).
// K tile staged async; V tile transposed through VGPRs into LDS.
// ---------------------------------------------------------------------------
__global__ __launch_bounds__(256)
void attn_wmma_kernel(const _Float16* __restrict__ q, const _Float16* __restrict__ k,
                      const _Float16* __restrict__ v, _Float16* __restrict__ out)
{
    constexpr int S = 1024, NH = 12, HD = 64, D = 768;
    int bid = blockIdx.x;
    int qb  = bid & 7;                // S / 128
    int hh  = (bid >> 3) % NH;
    int bb  = bid / (8 * NH);

    __shared__ __align__(16) _Float16 sK[32 * 64];    // [key][d]
    __shared__ __align__(16) _Float16 sVt[64 * 32];   // [d][key]
    __shared__ __align__(16) _Float16 sP[8][16 * 32]; // per-wave P repack

    int tid = threadIdx.x, lane = tid & 31, wave = tid >> 5;
    int la = lane & 15, lh = lane >> 4;
    int s0 = qb * 128 + wave * 16;

    // Q fragments, pre-scaled by 1/sqrt(HD) = 0.125
    v16h qf[2];
    #pragma unroll
    for (int c = 0; c < 2; ++c) {
        const _Float16* p = q + ((size_t)((bb * S + s0 + la) * NH + hh)) * HD + c * 32 + lh * 8;
        v8h lo = *(const v8h*)p;
        v8h hi = *(const v8h*)(p + 16);
        #pragma unroll
        for (int i = 0; i < 8; ++i) {
            qf[c][i]     = (_Float16)(lo[i] * (_Float16)0.125f);
            qf[c][i + 8] = (_Float16)(hi[i] * (_Float16)0.125f);
        }
    }

    v8f acc[4] = {};
    float mrow[8], lrow[8];
    #pragma unroll
    for (int i = 0; i < 8; ++i) { mrow[i] = -1e30f; lrow[i] = 0.f; }

    for (int jb = 0; jb < S; jb += 32) {
        __syncthreads();
        // stage K tile async; V tile transposed via VGPRs
        {
            int key = tid >> 3, ck = (tid & 7) * 8;
            size_t gofs = ((size_t)((bb * S + jb + key) * NH + hh)) * HD + ck;
            async_copy_b128(sK + key * 64 + ck, k + gofs);
            v8h vv = *(const v8h*)(v + gofs);
            #pragma unroll
            for (int i = 0; i < 8; ++i) sVt[(ck + i) * 32 + key] = vv[i];
        }
        async_wait0();
        __syncthreads();

        // scores: two 16x16 tiles (keys jb..+15, jb+16..+31), K-contraction = 64
        v8f sc[2] = {};
        #pragma unroll
        for (int t = 0; t < 2; ++t) {
            #pragma unroll
            for (int c = 0; c < 2; ++c) {
                const _Float16* p = sK + (t * 16 + la) * 64 + c * 32 + lh * 8;
                v8h lo = *(const v8h*)p;
                v8h hi = *(const v8h*)(p + 16);
                v16h bf;
                #pragma unroll
                for (int i = 0; i < 8; ++i) { bf[i] = lo[i]; bf[i + 8] = hi[i]; }
                sc[t] = __builtin_amdgcn_wmma_f32_16x16x32_f16(
                    false, qf[c], false, bf, (short)0, sc[t], false, false);
            }
        }

        // online softmax per row slot (row = i + 8*lh, cols across 16-lane group)
        #pragma unroll
        for (int i = 0; i < 8; ++i) {
            float tmax = fmaxf(sc[0][i], sc[1][i]);
            #pragma unroll
            for (int off = 1; off < 16; off <<= 1) tmax = fmaxf(tmax, __shfl_xor(tmax, off, 32));
            float mn = fmaxf(mrow[i], tmax);
            float p0 = __expf(sc[0][i] - mn);
            float p1 = __expf(sc[1][i] - mn);
            float rs = p0 + p1;
            #pragma unroll
            for (int off = 1; off < 16; off <<= 1) rs += __shfl_xor(rs, off, 32);
            float corr = __expf(mrow[i] - mn);
            lrow[i] = lrow[i] * corr + rs;
            mrow[i] = mn;
            #pragma unroll
            for (int t4 = 0; t4 < 4; ++t4) acc[t4][i] *= corr;
            int row = i + 8 * lh;
            sP[wave][row * 32 + la]      = (_Float16)p0;
            sP[wave][row * 32 + 16 + la] = (_Float16)p1;
        }

        // P fragment (16x32, A-layout) from wave-private LDS (in-order DS)
        v16h pf;
        {
            const _Float16* p = &sP[wave][la * 32 + lh * 8];
            v8h lo = *(const v8h*)p;
            v8h hi = *(const v8h*)(p + 16);
            #pragma unroll
            for (int i = 0; i < 8; ++i) { pf[i] = lo[i]; pf[i + 8] = hi[i]; }
        }
        // acc += P * V  (4 output tiles of 16 head-dims)
        #pragma unroll
        for (int t = 0; t < 4; ++t) {
            const _Float16* p = sVt + (t * 16 + la) * 32 + lh * 8;
            v8h lo = *(const v8h*)p;
            v8h hi = *(const v8h*)(p + 16);
            v16h bf;
            #pragma unroll
            for (int i = 0; i < 8; ++i) { bf[i] = lo[i]; bf[i + 8] = hi[i]; }
            acc[t] = __builtin_amdgcn_wmma_f32_16x16x32_f16(
                false, pf, false, bf, (short)0, acc[t], false, false);
        }
    }

    // normalize and store: out[b, s, hh*64 + d]
    #pragma unroll
    for (int t = 0; t < 4; ++t) {
        #pragma unroll
        for (int i = 0; i < 8; ++i) {
            int row = s0 + i + 8 * lh;
            int d   = hh * HD + t * 16 + la;
            out[(size_t)(bb * S + row) * D + d] = (_Float16)(acc[t][i] / lrow[i]);
        }
    }
}

// ---------------------------------------------------------------------------
// Host orchestration: 12 layers, all on `stream` (graph-capture safe).
// ---------------------------------------------------------------------------
extern "C" void kernel_launch(void* const* d_in, const int* in_sizes, int n_in,
                              void* d_out, int out_size, void* d_ws, size_t ws_size,
                              hipStream_t stream)
{
    (void)in_sizes; (void)n_in; (void)out_size; (void)ws_size;
    const float* x   = (const float*)d_in[0];
    const float* wq  = (const float*)d_in[1];
    const float* wk  = (const float*)d_in[2];
    const float* wv  = (const float*)d_in[3];
    const float* wo  = (const float*)d_in[4];
    const float* w1  = (const float*)d_in[5];
    const float* w2  = (const float*)d_in[6];
    const float* anw = (const float*)d_in[7];
    const float* fnw = (const float*)d_in[8];
    float* out = (float*)d_out;

    const int B = 8, S = 1024, D = 768, H = 2048, L = 12;
    const int R = B * S;  // 8192 rows

    char* ws = (char*)d_ws;
    auto take = [&](size_t bytes) -> char* {
        char* p = ws;
        ws += (bytes + 255) & ~(size_t)255;
        return p;
    };
    float*    h   = (float*)take((size_t)R * D * sizeof(float));        // residual stream
    _Float16* xn  = (_Float16*)take((size_t)R * D * sizeof(_Float16)); // normalized acts
    _Float16* at  = (_Float16*)take((size_t)R * D * sizeof(_Float16)); // attention out
    _Float16* qh  = (_Float16*)take((size_t)R * D * sizeof(_Float16));
    _Float16* kh  = (_Float16*)take((size_t)R * D * sizeof(_Float16));
    _Float16* vh  = (_Float16*)take((size_t)R * D * sizeof(_Float16));
    _Float16* ff1 = qh;  // FFN hidden (R*H f16 = 32MB) reuses dead q/k/v region (36MB)

    hipMemcpyAsync(h, x, (size_t)R * D * sizeof(float), hipMemcpyDeviceToDevice, stream);

    dim3 blk(256);
    dim3 gD(D / BN, R / BM);   // N = 768 GEMMs
    dim3 gH(H / BN, R / BM);   // N = 2048 GEMM
    dim3 gAttn(8 * 12 * (S / 128));

    for (int l = 0; l < L; ++l) {
        rmsnorm_cast_kernel<<<R, blk, 0, stream>>>(h, anw + (size_t)l * D, xn, D);
        gemm_wmma_kernel<<<gD, blk, 0, stream>>>(xn, wq + (size_t)l * D * D, nullptr, nullptr, qh, R, D, D, 0);
        gemm_wmma_kernel<<<gD, blk, 0, stream>>>(xn, wk + (size_t)l * D * D, nullptr, nullptr, kh, R, D, D, 0);
        gemm_wmma_kernel<<<gD, blk, 0, stream>>>(xn, wv + (size_t)l * D * D, nullptr, nullptr, vh, R, D, D, 0);
        attn_wmma_kernel<<<gAttn, blk, 0, stream>>>(qh, kh, vh, at);
        gemm_wmma_kernel<<<gD, blk, 0, stream>>>(at, wo + (size_t)l * D * D, h, h, nullptr, R, D, D, 1);
        rmsnorm_cast_kernel<<<R, blk, 0, stream>>>(h, fnw + (size_t)l * D, xn, D);
        gemm_wmma_kernel<<<gH, blk, 0, stream>>>(xn, w1 + (size_t)l * H * D, nullptr, nullptr, ff1, R, H, D, 2);
        float* o2 = (l == L - 1) ? out : h;
        gemm_wmma_kernel<<<gD, blk, 0, stream>>>(ff1, w2 + (size_t)l * D * H, h, o2, nullptr, R, D, H, 1);
    }
}
